// MultiHeadAttention_81123342286996
// MI455X (gfx1250) — matrix operand order repaired and checked by reading, not compiled, not run on hardware
//
#include <hip/hip_runtime.h>
#include <hip/hip_bf16.h>

#define DEVI __device__ __forceinline__

typedef __attribute__((ext_vector_type(16))) __bf16 v16bf;
typedef __attribute__((ext_vector_type(8)))  float  v8f;
typedef __attribute__((ext_vector_type(8)))  unsigned int v8u;

static constexpr int BB = 4, SS = 2048, DM = 128, HH = 8, DI = 1024;
static constexpr int TT = BB * SS;   // 8192 tokens

// ---------- helpers ----------

DEVI unsigned short f32_to_bf16(float f) {
  unsigned int u = __float_as_uint(f);
  u += 0x7FFFu + ((u >> 16) & 1u);   // round-to-nearest-even
  return (unsigned short)(u >> 16);
}
DEVI unsigned int pack2_bf16(float a, float b) {
  return (unsigned int)f32_to_bf16(a) | ((unsigned int)f32_to_bf16(b) << 16);
}

// A-matrix 16x32 bf16 fragment (contiguous k order used here, not the ISA table order): lane holds row m=lane&15;
// VGPR j: K = 16*hi + 2j,2j+1
// -> two contiguous 16B loads at byte offsets hi*32 and hi*32+16 from the row.
DEVI v8u load_fragA(const unsigned short* p, int hi) {
  const uint4* q0 = (const uint4*)((const char*)p + hi * 32);
  const uint4* q1 = (const uint4*)((const char*)p + hi * 32 + 16);
  uint4 a = *q0, b = *q1;
  v8u f; f[0]=a.x; f[1]=a.y; f[2]=a.z; f[3]=a.w; f[4]=b.x; f[5]=b.y; f[6]=b.z; f[7]=b.w;
  return f;
}
// B-matrix 32x16 bf16 fragment: lane holds col n=lane&15;
// VGPR j: K = 16*hi + 2j,2j+1  (contiguous order, same as load_fragA)
// -> two contiguous 16B loads at byte offsets hi*32 and hi*32+16.
DEVI v8u load_fragB(const unsigned short* p, int hi) {
  const uint4* q0 = (const uint4*)((const char*)p + hi * 32);
  const uint4* q1 = (const uint4*)((const char*)p + hi * 32 + 16);
  uint4 a = *q0, b = *q1;
  v8u f; f[0]=a.x; f[1]=a.y; f[2]=a.z; f[3]=a.w; f[4]=b.x; f[5]=b.y; f[6]=b.z; f[7]=b.w;
  return f;
}
DEVI v8f wmma_bf16(v8u a, v8u b, v8f c) {
  return __builtin_amdgcn_wmma_f32_16x16x32_bf16(
      false, __builtin_bit_cast(v16bf, a),
      false, __builtin_bit_cast(v16bf, b),
      (short)0, c, false, false);
}

// ---------- stage 1: conversions ----------

__global__ __launch_bounds__(256) void conv_kernel(const float* __restrict__ in,
                                                   unsigned short* __restrict__ out, int n) {
  int i = blockIdx.x * 256 + threadIdx.x;
  if (i < n) out[i] = f32_to_bf16(in[i]);
}

// out[(c)*R + r] = bf16(in[r*C + c])  : transpose + convert
__global__ __launch_bounds__(256) void transp_kernel(const float* __restrict__ in,
                                                     unsigned short* __restrict__ out,
                                                     int R, int C) {
  int i = blockIdx.x * 256 + threadIdx.x;
  if (i < R * C) {
    int r = i / C, c = i % C;
    out[(size_t)c * R + r] = f32_to_bf16(in[i]);
  }
}

// ---------- stage 2: QKV projection + bias + RoPE ----------
// MODE 0: Q -> rope, layout (B,H,S,128)
// MODE 1: K -> rope, layout (B,H,S,128)
// MODE 2: V -> transposed layout (B,H,128,S)
template <int MODE>
__global__ __launch_bounds__(256) void qkv_kernel(const unsigned short* __restrict__ X,
                                                  const unsigned short* __restrict__ WT,
                                                  const float* __restrict__ bias,
                                                  unsigned short* __restrict__ out) {
  const int lane = threadIdx.x & 31;
  const int wave = threadIdx.x >> 5;
  const int hi = lane >> 4;
  const int ln = lane & 15;
  const int tt = blockIdx.x;                 // token tile (16 tokens)
  const int dt = blockIdx.y * 8 + wave;      // out-dim tile (16 dims), 0..63

  const unsigned short* Xrow = X + (size_t)(tt * 16 + ln) * DM;   // A row: token
  const unsigned short* Wrow = WT + (size_t)(dt * 16 + ln) * DM;  // B row: out-dim

  // preload all 8 fragments so loads batch into one clause with staggered waits
  v8u af[4], bf[4];
#pragma unroll
  for (int cc = 0; cc < 4; cc++) {
    af[cc] = load_fragA(Xrow + cc * 32, hi);
    bf[cc] = load_fragB(Wrow + cc * 32, hi);
  }
  v8f c = {};
#pragma unroll
  for (int cc = 0; cc < 4; cc++) c = wmma_bf16(af[cc], bf[cc], c);

  const int n = dt * 16 + ln;         // global out dim in [0,1024)
  const float bv = bias[n];
#pragma unroll
  for (int i = 0; i < 8; i++) c[i] += bv;

  const int h = n >> 7;
  const int dd = n & 127;

  if constexpr (MODE < 2) {
    // RoPE: pairs (2p,2p+1); partner value lives in lane^1 (same rows).
    const int p = dd >> 1;
    const float inv_freq = __expf(-((float)(2 * p) / 128.0f) * 9.21034037198f); // ln(10000)
    const bool odd = (dd & 1) != 0;
#pragma unroll
    for (int i = 0; i < 8; i++) {
      int tok = tt * 16 + i + 8 * hi;
      int s = tok & (SS - 1);
      int b = tok >> 11;
      float ang = (float)s * inv_freq;
      float cs = __cosf(ang), sn = __sinf(ang);
      float partner = __shfl_xor(c[i], 1, 32);
      float r = odd ? (c[i] * cs + partner * sn) : (c[i] * cs - partner * sn);
      size_t addr = (((size_t)(b * HH + h)) * SS + s) * DM + dd;
      out[addr] = f32_to_bf16(r);
    }
  } else {
    // V transposed (B,H,128,S): lane's 8 rows are 8 consecutive s -> one b128 store
    int tok0 = tt * 16 + 8 * hi;
    int s0 = tok0 & (SS - 1);
    int b = tok0 >> 11;
    size_t base = (((size_t)(b * HH + h)) * DM + dd) * SS + s0;
    uint4 o;
    o.x = pack2_bf16(c[0], c[1]);
    o.y = pack2_bf16(c[2], c[3]);
    o.z = pack2_bf16(c[4], c[5]);
    o.w = pack2_bf16(c[6], c[7]);
    *(uint4*)(out + base) = o;
  }
}

// ---------- stage 3: flash attention (S^T formulation) ----------
__global__ __launch_bounds__(256) void fa_kernel(const unsigned short* __restrict__ Q,
                                                 const unsigned short* __restrict__ K,
                                                 const unsigned short* __restrict__ Vt,
                                                 unsigned short* __restrict__ ctx) {
  const int lane = threadIdx.x & 31;
  const int wave = threadIdx.x >> 5;
  const int hi = lane >> 4;
  const int lq = lane & 15;

  const int tile = blockIdx.x * 8 + wave;   // 0..4095
  const int QT = SS / 16;                   // 128
  const int bh = tile / QT;                 // 0..31
  const int qt = tile % QT;
  const int b = bh >> 3, h = bh & 7;
  const int q0 = qt * 16;

  const size_t head = (size_t)bh * SS * DM;
  const size_t headV = (size_t)bh * DM * SS;

  // Q as B-fragments (d x q), 4 chunks of K-dim 128, resident in VGPRs
  const unsigned short* Qrow = Q + head + (size_t)(q0 + lq) * DM;
  v8u qf[4];
#pragma unroll
  for (int c = 0; c < 4; c++) qf[c] = load_fragB(Qrow + c * 32, hi);

  v8f acc[8];
#pragma unroll
  for (int d = 0; d < 8; d++) acc[d] = v8f{};
  float m = -1e30f, l = 0.0f;

  const float scale = 0.08838834764831845f;  // 1/sqrt(128)
  const int qg = q0 + lq;
  const int nkc = (q0 + 16 + 31) >> 5;       // causal: key chunks of 32

  for (int kc = 0; kc < nkc; kc++) {
    const int k0 = kc * 32;
    const unsigned short* Krow0 = K + head + (size_t)(k0 + lq) * DM;
    const unsigned short* Krow1 = Krow0 + 16 * DM;
    const unsigned short* Vbase = Vt + headV + (size_t)lq * SS + k0;

    if (kc + 1 < nkc) {                      // prefetch next key/value chunk
      __builtin_prefetch(Krow0 + 32 * DM, 0, 1);
      __builtin_prefetch(Krow1 + 32 * DM, 0, 1);
      __builtin_prefetch(Vbase + 32, 0, 1);
      __builtin_prefetch(Vbase + (size_t)64 * SS + 32, 0, 1);
    }

    // --- preload all 8 K fragments (one clause, staggered waits) ---
    v8u ka[4], kb[4];
#pragma unroll
    for (int c = 0; c < 4; c++) {
      ka[c] = load_fragA(Krow0 + c * 32, hi);
      kb[c] = load_fragA(Krow1 + c * 32, hi);
    }
    // S^T tiles: A = K rows (keys x d), B = Q (d x queries)
    v8f s0 = {}, s1 = {};
#pragma unroll
    for (int c = 0; c < 4; c++) {
      s0 = wmma_bf16(ka[c], qf[c], s0);
      s1 = wmma_bf16(kb[c], qf[c], s1);
    }

    // --- issue V fragment loads early; softmax VALU work hides their latency ---
    v8u vf[8];
#pragma unroll
    for (int d = 0; d < 8; d++)
      vf[d] = load_fragA(Vbase + (size_t)(d * 16) * SS, hi);

    // scale + causal mask; rows: key = k0 + i + 8*hi (tile0), +16 (tile1)
    float p0[8], p1[8];
    float mc = -1e30f;
#pragma unroll
    for (int i = 0; i < 8; i++) {
      int kg = k0 + i + 8 * hi;
      float v0 = s0[i] * scale; if (kg      > qg) v0 = -1e30f;
      float v1 = s1[i] * scale; if (kg + 16 > qg) v1 = -1e30f;
      p0[i] = v0; p1[i] = v1;
      mc = fmaxf(mc, fmaxf(v0, v1));
    }
    mc = fmaxf(mc, __shfl_xor(mc, 16, 32));
    float mnew = fmaxf(m, mc);
    float alpha = __expf(m - mnew);
    m = mnew;

    float lsum = 0.0f;
#pragma unroll
    for (int i = 0; i < 8; i++) {
      p0[i] = __expf(p0[i] - mnew);
      p1[i] = __expf(p1[i] - mnew);
      lsum += p0[i] + p1[i];
    }
    lsum += __shfl_xor(lsum, 16, 32);
    l = l * alpha + lsum;
#pragma unroll
    for (int d = 0; d < 8; d++)
#pragma unroll
      for (int i = 0; i < 8; i++) acc[d][i] *= alpha;

    // Build P^T as B-fragment (32 keys x 16 queries).
    // B needs: hi=0 -> tile0 rows 0..15 ; hi=1 -> tile1 rows 16..31.
    // Own half holds rows (8*hi..8*hi+7); the other 8 rows come from lane^16.
    float recv[8];
#pragma unroll
    for (int i = 0; i < 8; i++)
      recv[i] = __shfl_xor(hi ? p0[i] : p1[i], 16, 32);

    v8u pf;
    if (hi == 0) {
#pragma unroll
      for (int j = 0; j < 4; j++) pf[j]     = pack2_bf16(p0[2 * j], p0[2 * j + 1]);
#pragma unroll
      for (int j = 0; j < 4; j++) pf[4 + j] = pack2_bf16(recv[2 * j], recv[2 * j + 1]);
    } else {
#pragma unroll
      for (int j = 0; j < 4; j++) pf[j]     = pack2_bf16(recv[2 * j], recv[2 * j + 1]);
#pragma unroll
      for (int j = 0; j < 4; j++) pf[4 + j] = pack2_bf16(p1[2 * j], p1[2 * j + 1]);
    }

    // O^T += V^T(16d x 32k) * P^T(32k x 16q), for 8 d-tiles
#pragma unroll
    for (int d = 0; d < 8; d++)
      acc[d] = wmma_bf16(vf[d], pf, acc[d]);
  }

  // normalize + store ctx (B,S,1024) bf16; lane's rows = 8 consecutive d
  const float inv = 1.0f / l;
  const size_t crow = ((size_t)(b * SS + q0 + lq)) * DI + h * DM;
#pragma unroll
  for (int d = 0; d < 8; d++) {
    uint4 o;
    o.x = pack2_bf16(acc[d][0] * inv, acc[d][1] * inv);
    o.y = pack2_bf16(acc[d][2] * inv, acc[d][3] * inv);
    o.z = pack2_bf16(acc[d][4] * inv, acc[d][5] * inv);
    o.w = pack2_bf16(acc[d][6] * inv, acc[d][7] * inv);
    *(uint4*)(ctx + crow + d * 16 + hi * 8) = o;
  }
}

// ---------- stage 4: output projection ----------
__global__ __launch_bounds__(256) void proj_kernel(const unsigned short* __restrict__ ctx,
                                                   const unsigned short* __restrict__ WoT,
                                                   const float* __restrict__ bo,
                                                   float* __restrict__ out) {
  const int lane = threadIdx.x & 31;
  const int wave = threadIdx.x >> 5;   // dim tile 0..7
  const int hi = lane >> 4;
  const int ln = lane & 15;
  const int tt = blockIdx.x;

  const unsigned short* Arow = ctx + (size_t)(tt * 16 + ln) * DI;
  const unsigned short* Brow = WoT + (size_t)(wave * 16 + ln) * DI;

  v8f c = {};
  // step by 4 chunks: 8 fragments (16 b128 loads) in flight per step
  for (int cc = 0; cc < 32; cc += 4) {
    v8u af[4], bf[4];
#pragma unroll
    for (int u = 0; u < 4; u++) {
      af[u] = load_fragA(Arow + (cc + u) * 32, hi);
      bf[u] = load_fragB(Brow + (cc + u) * 32, hi);
    }
#pragma unroll
    for (int u = 0; u < 4; u++) c = wmma_bf16(af[u], bf[u], c);
  }
  const int n = wave * 16 + ln;
  const float bv = bo[n];
#pragma unroll
  for (int i = 0; i < 8; i++) {
    int tok = tt * 16 + i + 8 * hi;
    out[(size_t)tok * DM + n] = c[i] + bv;
  }
}

// ---------- launch ----------

extern "C" void kernel_launch(void* const* d_in, const int* in_sizes, int n_in,
                              void* d_out, int out_size, void* d_ws, size_t ws_size,
                              hipStream_t stream) {
  (void)in_sizes; (void)n_in; (void)out_size; (void)ws_size;

  const float* query = (const float*)d_in[0];
  const float* key   = (const float*)d_in[1];
  const float* value = (const float*)d_in[2];
  // d_in[3] = mask (causal, handled analytically)
  const float* Wq = (const float*)d_in[4];
  const float* bq = (const float*)d_in[5];
  const float* Wk = (const float*)d_in[6];
  const float* bk = (const float*)d_in[7];
  const float* Wv = (const float*)d_in[8];
  const float* bv = (const float*)d_in[9];
  const float* Wo = (const float*)d_in[10];
  const float* bo = (const float*)d_in[11];
  float* out = (float*)d_out;

  unsigned short* w = (unsigned short*)d_ws;
  size_t off = 0;
  unsigned short* Xq  = w + off; off += (size_t)TT * DM;
  unsigned short* Xk  = w + off; off += (size_t)TT * DM;
  unsigned short* Xv  = w + off; off += (size_t)TT * DM;
  unsigned short* WqT = w + off; off += (size_t)DI * DM;
  unsigned short* WkT = w + off; off += (size_t)DI * DM;
  unsigned short* WvT = w + off; off += (size_t)DI * DM;
  unsigned short* WoT = w + off; off += (size_t)DM * DI;
  unsigned short* Qb  = w + off; off += (size_t)BB * HH * SS * DM;
  unsigned short* Kb  = w + off; off += (size_t)BB * HH * SS * DM;
  unsigned short* Vt  = w + off; off += (size_t)BB * HH * DM * SS;
  unsigned short* ctx = w + off; off += (size_t)TT * DI;

  const int nx = TT * DM;
  conv_kernel<<<(nx + 255) / 256, 256, 0, stream>>>(query, Xq, nx);
  conv_kernel<<<(nx + 255) / 256, 256, 0, stream>>>(key,   Xk, nx);
  conv_kernel<<<(nx + 255) / 256, 256, 0, stream>>>(value, Xv, nx);

  const int nw = DM * DI;
  transp_kernel<<<(nw + 255) / 256, 256, 0, stream>>>(Wq, WqT, DM, DI);
  transp_kernel<<<(nw + 255) / 256, 256, 0, stream>>>(Wk, WkT, DM, DI);
  transp_kernel<<<(nw + 255) / 256, 256, 0, stream>>>(Wv, WvT, DM, DI);
  transp_kernel<<<(nw + 255) / 256, 256, 0, stream>>>(Wo, WoT, DI, DM);

  dim3 gq(TT / 16, 8);
  qkv_kernel<0><<<gq, 256, 0, stream>>>(Xq, WqT, bq, Qb);
  qkv_kernel<1><<<gq, 256, 0, stream>>>(Xk, WkT, bk, Kb);
  qkv_kernel<2><<<gq, 256, 0, stream>>>(Xv, WvT, bv, Vt);

  fa_kernel<<<(BB * HH * (SS / 16)) / 8, 256, 0, stream>>>(Qb, Kb, Vt, ctx);

  proj_kernel<<<TT / 16, 256, 0, stream>>>(ctx, WoT, bo, out);
}
